// Pointnet2MSG_seg_feature_26164940767539
// MI455X (gfx1250) — compile-verified, matrix operand order checked by reading, not run on hardware
//
#include <hip/hip_runtime.h>

// ---------------------------------------------------------------------------
// PointNet++ MSG segmentation feature pipeline for gfx1250 (MI455X).
// f16 activations/weights, f32 WMMA accumulation (v_wmma_f32_16x16x32_f16),
// f32 batch-norm statistics. Memory-bound workload (~1.5GB traffic, ~100GFLOP)
// so f16 storage + WMMA is the right operating point.
// ---------------------------------------------------------------------------

typedef _Float16 h16;
typedef _Float16 half8 __attribute__((ext_vector_type(8)));
typedef _Float16 v16h  __attribute__((ext_vector_type(16)));
typedef float    v8f   __attribute__((ext_vector_type(8)));

#define BATCH 16
#define NPTS  8192

static __device__ __forceinline__ size_t gtid() {
  return (size_t)blockIdx.x * blockDim.x + threadIdx.x;
}
static __device__ __forceinline__ size_t gstride() {
  return (size_t)gridDim.x * blockDim.x;
}
static __device__ __forceinline__ v16h cat16(half8 lo, half8 hi) {
  return __builtin_shufflevector(lo, hi, 0, 1, 2, 3, 4, 5, 6, 7,
                                 8, 9, 10, 11, 12, 13, 14, 15);
}

// ---------------------------------------------------------------------------
// split pc -> xyz (f32) + normals (f16)
__global__ void k_split(const float* __restrict__ pc, float* __restrict__ xyz,
                        h16* __restrict__ nor) {
  size_t total = (size_t)BATCH * NPTS;
  for (size_t g = gtid(); g < total; g += gstride()) {
    const float* p = pc + g * 6;
    xyz[g * 3 + 0] = p[0]; xyz[g * 3 + 1] = p[1]; xyz[g * 3 + 2] = p[2];
    nor[g * 3 + 0] = (h16)p[3]; nor[g * 3 + 1] = (h16)p[4]; nor[g * 3 + 2] = (h16)p[5];
  }
}

// ---------------------------------------------------------------------------
// Farthest point sampling: one block per batch. N = PER*256.
template <int PER>
__global__ void k_fps(const float* __restrict__ xyz, int npoint, int* __restrict__ out) {
  const int N = PER * 256;
  int b = blockIdx.x;
  const float* px = xyz + (size_t)b * N * 3;
  int* po = out + (size_t)b * npoint;
  __shared__ float sv[256];
  __shared__ int   si[256];
  __shared__ float sp[3];
  float mind[PER];
  int t = threadIdx.x;
#pragma unroll
  for (int i = 0; i < PER; ++i) mind[i] = 1e10f;
  if (t == 0) { po[0] = 0; sp[0] = px[0]; sp[1] = px[1]; sp[2] = px[2]; }
  __syncthreads();
  for (int it = 1; it < npoint; ++it) {
    float cx = sp[0], cy = sp[1], cz = sp[2];
    float best = -1.f; int bi = 0;
#pragma unroll
    for (int i = 0; i < PER; ++i) {
      int p = t + i * 256;
      float dx = px[p * 3] - cx, dy = px[p * 3 + 1] - cy, dz = px[p * 3 + 2] - cz;
      float d = dx * dx + dy * dy + dz * dz;
      if (d < mind[i]) mind[i] = d;
      if (mind[i] > best) { best = mind[i]; bi = p; }
    }
    sv[t] = best; si[t] = bi;
    __syncthreads();
    for (int s = 128; s > 0; s >>= 1) {
      if (t < s) {
        if (sv[t + s] > sv[t] || (sv[t + s] == sv[t] && si[t + s] < si[t])) {
          sv[t] = sv[t + s]; si[t] = si[t + s];
        }
      }
      __syncthreads();
    }
    if (t == 0) {
      int sel = si[0];
      po[it] = sel;
      sp[0] = px[sel * 3]; sp[1] = px[sel * 3 + 1]; sp[2] = px[sel * 3 + 2];
    }
    __syncthreads();
  }
}

// gather new_xyz = xyz[idx]
__global__ void k_gather3(const float* __restrict__ xyz, const int* __restrict__ idx,
                          int N, int np, float* __restrict__ out) {
  size_t total = (size_t)BATCH * np;
  for (size_t g = gtid(); g < total; g += gstride()) {
    int b = (int)(g / np);
    int p = idx[g];
    out[g * 3 + 0] = xyz[((size_t)b * N + p) * 3 + 0];
    out[g * 3 + 1] = xyz[((size_t)b * N + p) * 3 + 1];
    out[g * 3 + 2] = xyz[((size_t)b * N + p) * 3 + 2];
  }
}

// ---------------------------------------------------------------------------
// Ball query: one wave per query point; ordered append via ballot prefix.
__global__ void k_ballq(const float* __restrict__ xyz, const float* __restrict__ nxyz,
                        int N, int np, float r2, int ns, int* __restrict__ idx,
                        int total_q) {
  int wpb = blockDim.x >> 5;
  int gw = blockIdx.x * wpb + (threadIdx.x >> 5);
  int lane = threadIdx.x & 31;
  if (gw >= total_q) return;
  int b = gw / np;
  const float* q = nxyz + (size_t)gw * 3;
  float qx = q[0], qy = q[1], qz = q[2];
  const float* px = xyz + (size_t)b * N * 3;
  int* out = idx + (size_t)gw * ns;
  int filled = 0;
  int first = 0x7fffffff;
  for (int base = 0; base < N && filled < ns; base += 32) {
    int p = base + lane;
    bool ok = false;
    if (p < N) {
      float dx = px[p * 3] - qx, dy = px[p * 3 + 1] - qy, dz = px[p * 3 + 2] - qz;
      ok = (dx * dx + dy * dy + dz * dz) <= r2;
    }
    unsigned mask = (unsigned)__ballot(ok);
    int pos = filled + __popc(mask & ((1u << lane) - 1u));
    if (ok && pos < ns) {
      out[pos] = p;
      if (pos == 0) first = p;
    }
    filled += __popc(mask);
  }
  for (int off = 16; off > 0; off >>= 1) {
    int o = __shfl_xor(first, off, 32);
    if (o < first) first = o;
  }
  if (first == 0x7fffffff) first = 0;
  if (filled < ns)
    for (int pos = filled + lane; pos < ns; pos += 32) out[pos] = first;
}

// ---------------------------------------------------------------------------
// Build grouped input rows: [rel_xyz(3) | feats(Cf) | zero-pad] (f16, pitch Kpad)
__global__ void k_group(const float* __restrict__ xyz, const float* __restrict__ nxyz,
                        const h16* __restrict__ feats, int Cf,
                        const int* __restrict__ idx, int N, int np, int ns, int Kpad,
                        h16* __restrict__ A) {
  size_t total = (size_t)BATCH * np * ns * Kpad;
  for (size_t g = gtid(); g < total; g += gstride()) {
    int c = (int)(g % Kpad);
    size_t row = g / Kpad;
    int s = (int)(row % ns);
    size_t bj = row / ns;
    int j = (int)(bj % np);
    int b = (int)(bj / np);
    int p = idx[(((size_t)b * np + j) * ns) + s];
    float v = 0.f;
    if (c < 3)
      v = xyz[((size_t)b * N + p) * 3 + c] - nxyz[((size_t)b * np + j) * 3 + c];
    else if (c < 3 + Cf)
      v = (float)feats[((size_t)b * N + p) * Cf + (c - 3)];
    A[g] = (h16)v;
  }
}

// ---------------------------------------------------------------------------
// Repack weight f32[K,Nc] -> f16 B-fragments [(kb*NT+nt)*32 + lane][16]
__global__ void k_repack(const float* __restrict__ W, int K, int Nc, int Kpad, int Npad,
                         h16* __restrict__ Wr) {
  int NT = Npad >> 4;
  size_t total = (size_t)Kpad * Npad;
  for (size_t g = gtid(); g < total; g += gstride()) {
    size_t frag = g >> 9;        // /512
    int r = (int)(g & 511);
    int lane = r >> 4, j = r & 15;
    int kb = (int)(frag / NT), nt = (int)(frag % NT);
    int khalf = lane >> 4, col = lane & 15;
    int k = kb * 32 + ((j < 8) ? (khalf * 8 + j) : (8 + khalf * 8 + j));
    int n = nt * 16 + col;
    float v = (k < K && n < Nc) ? W[(size_t)k * Nc + n] : 0.f;
    Wr[g] = (h16)v;
  }
}

// ---------------------------------------------------------------------------
// WMMA GEMM: Y[M,Npad](f16) = A[M,Kpad](f16) x Wr (f32 accumulate).
// Block = 8 waves; each wave computes a 16 x (TILES*16) strip with TILES
// compile-time accumulators. All B fragments are loaded into distinct
// registers first so the loads issue as one clause and each WMMA only waits
// for its own loads (in-order loadcnt), overlapping math with later loads.
template <int TILES>
__global__ void __launch_bounds__(256)
k_gemm(const h16* __restrict__ A, const h16* __restrict__ Wr,
       h16* __restrict__ Y, int M, int Kpad, int Npad, int nOff) {
  const int NT = Npad >> 4;
  const int KB = Kpad >> 5;
  int wave = threadIdx.x >> 5;
  int lane = threadIdx.x & 31;
  int m0 = (blockIdx.x * 8 + wave) * 16;
  int n0 = blockIdx.y * (TILES * 16) + nOff;
  int ntBase = n0 >> 4;
  int row = m0 + (lane & 15);
  int khalf = lane >> 4;
  v8f acc[TILES] = {};
  const h16* ap = A + (size_t)row * Kpad + khalf * 8;
  const h16* bp = Wr + ((size_t)ntBase * 32 + lane) * 16;
  const size_t bstride = (size_t)NT * 512;
  for (int kb = 0; kb < KB; ++kb) {
    v16h av = cat16(*(const half8*)(ap), *(const half8*)(ap + 16));
    __builtin_prefetch(ap + 32, 0, 3);   // global_prefetch_b8 (speculative)
    v16h bv[TILES];
#pragma unroll
    for (int t = 0; t < TILES; ++t)
      bv[t] = cat16(*(const half8*)(bp + t * 512),
                    *(const half8*)(bp + t * 512 + 8));
#pragma unroll
    for (int t = 0; t < TILES; ++t)
      acc[t] = __builtin_amdgcn_wmma_f32_16x16x32_f16(
          false, av, false, bv[t], (short)0, acc[t], false, false);
    ap += 32;
    bp += bstride;
  }
  int col = lane & 15, rhalf = lane >> 4;
#pragma unroll
  for (int t = 0; t < TILES; ++t) {
    int n = n0 + t * 16 + col;
#pragma unroll
    for (int i = 0; i < 8; ++i)
      Y[(size_t)(m0 + rhalf * 8 + i) * Npad + n] = (h16)acc[t][i];
  }
}

// ---------------------------------------------------------------------------
// BN statistics: per-channel sum/sumsq via register->LDS->global reduction.
__global__ void k_bn_stats(const h16* __restrict__ Y, int M, int pitch, int Cout,
                           float* __restrict__ sum, float* __restrict__ sq) {
  __shared__ float ss[1024];
  __shared__ float sr[1024];
  for (int c = threadIdx.x; c < Cout; c += blockDim.x) { ss[c] = 0.f; sr[c] = 0.f; }
  __syncthreads();
  int r0 = blockIdx.x * 256;
  int r1 = r0 + 256; if (r1 > M) r1 = M;
  float ls[4] = {0.f, 0.f, 0.f, 0.f}, lq[4] = {0.f, 0.f, 0.f, 0.f};
  int nst = (Cout + blockDim.x - 1) / blockDim.x;   // <= 4
  for (int r = r0; r < r1; ++r) {
    const h16* rowp = Y + (size_t)r * pitch;
    for (int i = 0; i < nst; ++i) {
      int c = threadIdx.x + i * blockDim.x;
      if (c < Cout) { float x = (float)rowp[c]; ls[i] += x; lq[i] += x * x; }
    }
  }
  for (int i = 0; i < nst; ++i) {
    int c = threadIdx.x + i * blockDim.x;
    if (c < Cout) { atomicAdd(&ss[c], ls[i]); atomicAdd(&sr[c], lq[i]); }
  }
  __syncthreads();
  for (int c = threadIdx.x; c < Cout; c += blockDim.x) {
    atomicAdd(&sum[c], ss[c]);
    atomicAdd(&sq[c], sr[c]);
  }
}

// BN + ReLU in place (scale/shift recomputed per element from cached stats).
__global__ void k_bn_relu(h16* __restrict__ Y, size_t M, int pitch, int Cout,
                          const float* __restrict__ sum, const float* __restrict__ sq,
                          const float* __restrict__ gamma, const float* __restrict__ beta,
                          float invM) {
  size_t total = M * (size_t)pitch;
  for (size_t g = gtid(); g < total; g += gstride()) {
    int c = (int)(g % pitch);
    if (c >= Cout) { Y[g] = (h16)0.f; continue; }
    float mu = sum[c] * invM;
    float var = sq[c] * invM - mu * mu;
    float sc = gamma[c] * rsqrtf(var + 1e-5f);
    float sh = beta[c] - mu * sc;
    float x = (float)Y[g] * sc + sh;
    Y[g] = (h16)fmaxf(x, 0.f);
  }
}

// ---------------------------------------------------------------------------
// Max-pool over sample dim; write into concatenated feature buffer.
__global__ void k_maxpool(const h16* __restrict__ Y, int pitch, int Cout, int ns,
                          h16* __restrict__ dst, int dstPitch, int coff, int rows) {
  size_t total = (size_t)rows * Cout;
  for (size_t g = gtid(); g < total; g += gstride()) {
    size_t bp = g / Cout;
    int c = (int)(g % Cout);
    const h16* base = Y + (size_t)bp * ns * pitch + c;
    float m = -3e38f;
    for (int s = 0; s < ns; ++s) m = fmaxf(m, (float)base[(size_t)s * pitch]);
    dst[bp * dstPitch + coff + c] = (h16)m;
  }
}

// ---------------------------------------------------------------------------
// sa input: [l2_feat(512) | l2_xyz(3) | pad] -> A pitch 544
__global__ void k_sacat(const h16* __restrict__ f2, const float* __restrict__ l2x,
                        h16* __restrict__ A) {
  size_t total = (size_t)BATCH * 128 * 544;
  for (size_t g = gtid(); g < total; g += gstride()) {
    size_t row = g / 544;
    int c = (int)(g % 544);
    float v = 0.f;
    if (c < 512) v = (float)f2[row * 512 + c];
    else if (c < 515) v = l2x[row * 3 + (c - 512)];
    A[g] = (h16)v;
  }
}

// l0 features: [one_hot(16) | xyz(3) | normals(3)] pitch 22
__global__ void k_l0(const float* __restrict__ oh, const float* __restrict__ xyz,
                     const h16* __restrict__ nor, h16* __restrict__ l0) {
  size_t total = (size_t)BATCH * NPTS * 22;
  for (size_t g = gtid(); g < total; g += gstride()) {
    int c = (int)(g % 22);
    size_t bn = g / 22;
    int n = (int)(bn % NPTS);
    int b = (int)(bn / NPTS);
    float v;
    if (c < 16) v = oh[b * 16 + c];
    else if (c < 19) v = xyz[((size_t)b * NPTS + n) * 3 + (c - 16)];
    else v = (float)nor[((size_t)b * NPTS + n) * 3 + (c - 19)];
    l0[g] = (h16)v;
  }
}

// ---------------------------------------------------------------------------
// Feature propagation: 3-NN inverse-distance interpolation + concat unknown
// feats -> A rows [interp(Ck) | ufeat(Cu) | pad]. One wave per unknown point,
// known xyz staged in LDS.
__global__ void k_fp(const float* __restrict__ kxyz, const float* __restrict__ uxyz,
                     const h16* __restrict__ kfeat, int Ck, int kPitch,
                     const h16* __restrict__ ufeat, int Cu, int uPitch,
                     int nk, int nu, int Kpad, h16* __restrict__ A, int chunks) {
  __shared__ float s_k[512 * 3];
  int b = blockIdx.x / chunks;
  int u0 = (blockIdx.x % chunks) * 8;
  for (int i = threadIdx.x; i < nk * 3; i += blockDim.x)
    s_k[i] = kxyz[(size_t)b * nk * 3 + i];
  __syncthreads();
  int wave = threadIdx.x >> 5, lane = threadIdx.x & 31;
  int u = u0 + wave;
  if (u >= nu) return;
  const float* up = uxyz + ((size_t)b * nu + u) * 3;
  float ux = up[0], uy = up[1], uz = up[2];
  float d0 = 3e38f, d1 = 3e38f, d2 = 3e38f;
  int i0 = 0, i1 = 0, i2 = 0;
  for (int kk = 0; kk < nk; ++kk) {
    float dx = s_k[kk * 3] - ux, dy = s_k[kk * 3 + 1] - uy, dz = s_k[kk * 3 + 2] - uz;
    float d = dx * dx + dy * dy + dz * dz;
    if (d < d0)      { d2 = d1; i2 = i1; d1 = d0; i1 = i0; d0 = d; i0 = kk; }
    else if (d < d1) { d2 = d1; i2 = i1; d1 = d; i1 = kk; }
    else if (d < d2) { d2 = d; i2 = kk; }
  }
  float w0 = 1.f / (d0 + 1e-8f), w1 = 1.f / (d1 + 1e-8f), w2 = 1.f / (d2 + 1e-8f);
  float wsum = w0 + w1 + w2;
  w0 /= wsum; w1 /= wsum; w2 /= wsum;
  const h16* k0 = kfeat + ((size_t)b * nk + i0) * kPitch;
  const h16* k1 = kfeat + ((size_t)b * nk + i1) * kPitch;
  const h16* k2 = kfeat + ((size_t)b * nk + i2) * kPitch;
  const h16* uf = ufeat + ((size_t)b * nu + u) * uPitch;
  h16* arow = A + ((size_t)b * nu + u) * Kpad;
  for (int c = lane; c < Kpad; c += 32) {
    float v = 0.f;
    if (c < Ck)
      v = w0 * (float)k0[c] + w1 * (float)k1[c] + w2 * (float)k2[c];
    else if (c < Ck + Cu)
      v = (float)uf[c - Ck];
    arow[c] = (h16)v;
  }
}

// ---------------------------------------------------------------------------
__global__ void k_zero(float* __restrict__ p, int n) {
  for (size_t i = gtid(); i < (size_t)n; i += gstride()) p[i] = 0.f;
}

// final transpose: up3 f16 [B*8192, 128] -> out f32 [B,128,8192]
__global__ void k_out(const h16* __restrict__ up3, float* __restrict__ out) {
  size_t total = (size_t)BATCH * 128 * NPTS;
  for (size_t g = gtid(); g < total; g += gstride()) {
    int n = (int)(g % NPTS);
    size_t bc = g / NPTS;
    int c = (int)(bc % 128);
    int b = (int)(bc / 128);
    out[g] = (float)up3[((size_t)b * NPTS + n) * 128 + c];
  }
}

// ---------------------------------------------------------------------------
static inline size_t alignUp(size_t x) { return (x + 255) & ~(size_t)255; }
static inline int rup32(int x) { return (x + 31) & ~31; }
static inline int gsBlocks(size_t total) {
  size_t b = (total + 255) / 256;
  return (int)(b > 4096 ? 4096 : (b == 0 ? 1 : b));
}

extern "C" void kernel_launch(void* const* d_in, const int* in_sizes, int n_in,
                              void* d_out, int out_size, void* d_ws, size_t ws_size,
                              hipStream_t stream) {
  (void)in_sizes; (void)n_in; (void)out_size; (void)ws_size;
  const float* pc = (const float*)d_in[0];
  const float* oneh = (const float*)d_in[1];

  // --- workspace bump allocator (~580 MB total) ---
  char* base = (char*)d_ws;
  size_t off = 0;
  auto alloc = [&](size_t bytes) -> void* {
    void* p = base + off;
    off = alignUp(off + bytes);
    return p;
  };
  float* f_xyz   = (float*)alloc((size_t)BATCH * NPTS * 3 * 4);
  h16*   h_norm  = (h16*)  alloc((size_t)BATCH * NPTS * 3 * 2);
  h16*   h_l0    = (h16*)  alloc((size_t)BATCH * NPTS * 22 * 2);
  int*   i_fps1  = (int*)  alloc((size_t)BATCH * 512 * 4);
  float* f_l1x   = (float*)alloc((size_t)BATCH * 512 * 3 * 4);
  int*   i_fps2  = (int*)  alloc((size_t)BATCH * 128 * 4);
  float* f_l2x   = (float*)alloc((size_t)BATCH * 128 * 3 * 4);
  int*   i_ball  = (int*)  alloc((size_t)BATCH * 512 * 128 * 4);
  h16*   h_f1    = (h16*)  alloc((size_t)BATCH * 512 * 320 * 2);
  h16*   h_f2    = (h16*)  alloc((size_t)BATCH * 128 * 512 * 2);
  h16*   h_l3    = (h16*)  alloc((size_t)2048 * 1024 * 2);
  h16*   h_up1   = (h16*)  alloc((size_t)2048 * 256 * 2);
  h16*   h_up2   = (h16*)  alloc((size_t)8192 * 128 * 2);
  h16*   h_up3   = (h16*)  alloc((size_t)131072 * 128 * 2);
  h16*   h_wr    = (h16*)  alloc((size_t)2 * 1024 * 1024);
  float* f_sum   = (float*)alloc((size_t)2048 * 4);
  float* f_sq    = f_sum + 1024;
  h16*   big0    = (h16*)  alloc((size_t)268435456);
  h16*   big1    = (h16*)  alloc((size_t)268435456);

  auto runLayer = [&](const h16* Ab, h16* Yb, int M, int K, int Nc,
                      int wi, int bi, int gi) {
    int Kp = rup32(K), Np = rup32(Nc);
    k_repack<<<gsBlocks((size_t)Kp * Np), 256, 0, stream>>>(
        (const float*)d_in[wi], K, Nc, Kp, Np, h_wr);
    k_zero<<<1, 256, 0, stream>>>(f_sum, 2048);
    int nt64 = Np / 64;
    if (nt64 > 0)
      k_gemm<4><<<dim3(M / 128, nt64), 256, 0, stream>>>(Ab, h_wr, Yb, M, Kp, Np, 0);
    if (Np % 64)   // 32-wide remainder (Np is always a multiple of 32)
      k_gemm<2><<<dim3(M / 128, 1), 256, 0, stream>>>(Ab, h_wr, Yb, M, Kp, Np, nt64 * 64);
    k_bn_stats<<<(M + 255) / 256, 256, 0, stream>>>(Yb, M, Np, Nc, f_sum, f_sq);
    k_bn_relu<<<gsBlocks((size_t)M * Np), 256, 0, stream>>>(
        Yb, (size_t)M, Np, Nc, f_sum, f_sq,
        (const float*)d_in[gi], (const float*)d_in[bi], 1.0f / (float)M);
  };

  // --- stage 0: split ---
  k_split<<<gsBlocks((size_t)BATCH * NPTS), 256, 0, stream>>>(pc, f_xyz, h_norm);

  // --- msg1: npoint=512 over N=8192, feats = normals (3ch) ---
  k_fps<32><<<BATCH, 256, 0, stream>>>(f_xyz, 512, i_fps1);
  k_gather3<<<gsBlocks((size_t)BATCH * 512), 256, 0, stream>>>(f_xyz, i_fps1, NPTS, 512, f_l1x);

  struct Br1 { float r; int ns; int c1, c2, c3; int w0, w1, w2; int coff; };
  const Br1 m1[3] = {
      {0.1f,  32, 32, 32,  64, 20, 23, 26, 0},
      {0.2f,  64, 64, 64, 128, 29, 32, 35, 64},
      {0.4f, 128, 64, 96, 128, 38, 41, 44, 192}};
  for (int br = 0; br < 3; ++br) {
    const Br1& L = m1[br];
    int M = BATCH * 512 * L.ns;
    int totq = BATCH * 512;
    k_ballq<<<(totq + 7) / 8, 256, 0, stream>>>(f_xyz, f_l1x, NPTS, 512,
                                                L.r * L.r, L.ns, i_ball, totq);
    k_group<<<gsBlocks((size_t)M * 32), 256, 0, stream>>>(
        f_xyz, f_l1x, h_norm, 3, i_ball, NPTS, 512, L.ns, 32, big0);
    runLayer(big0, big1, M, 6,    L.c1, L.w0, L.w0 + 1, L.w0 + 2);
    runLayer(big1, big0, M, L.c1, L.c2, L.w1, L.w1 + 1, L.w1 + 2);
    runLayer(big0, big1, M, L.c2, L.c3, L.w2, L.w2 + 1, L.w2 + 2);
    k_maxpool<<<gsBlocks((size_t)BATCH * 512 * L.c3), 256, 0, stream>>>(
        big1, rup32(L.c3), L.c3, L.ns, h_f1, 320, L.coff, BATCH * 512);
  }

  // --- msg2: npoint=128 over 512 pts, feats = l1 (320ch) ---
  k_fps<2><<<BATCH, 256, 0, stream>>>(f_l1x, 128, i_fps2);
  k_gather3<<<gsBlocks((size_t)BATCH * 128), 256, 0, stream>>>(f_l1x, i_fps2, 512, 128, f_l2x);

  struct Br2 { float r; int ns; int c1, c2, c3; int w0, w1, w2; int coff; };
  const Br2 m2[2] = {
      {0.4f,  64, 128, 128, 256, 47, 50, 53, 0},
      {0.8f, 128, 128, 196, 256, 56, 59, 62, 256}};
  for (int br = 0; br < 2; ++br) {
    const Br2& L = m2[br];
    int M = BATCH * 128 * L.ns;
    int totq = BATCH * 128;
    k_ballq<<<(totq + 7) / 8, 256, 0, stream>>>(f_l1x, f_l2x, 512, 128,
                                                L.r * L.r, L.ns, i_ball, totq);
    k_group<<<gsBlocks((size_t)M * 352), 256, 0, stream>>>(
        f_l1x, f_l2x, h_f1, 320, i_ball, 512, 128, L.ns, 352, big0);
    runLayer(big0, big1, M, 323,  L.c1, L.w0, L.w0 + 1, L.w0 + 2);
    runLayer(big1, big0, M, L.c1, L.c2, L.w1, L.w1 + 1, L.w1 + 2);
    runLayer(big0, big1, M, L.c2, L.c3, L.w2, L.w2 + 1, L.w2 + 2);
    k_maxpool<<<gsBlocks((size_t)BATCH * 128 * L.c3), 256, 0, stream>>>(
        big1, rup32(L.c3), L.c3, L.ns, h_f2, 512, L.coff, BATCH * 128);
  }

  // --- sa: [515 -> 256 -> 512 -> 1024] on 2048 rows ---
  k_sacat<<<gsBlocks((size_t)2048 * 544), 256, 0, stream>>>(h_f2, f_l2x, big0);
  runLayer(big0, big1, 2048, 515, 256, 65, 66, 67);
  runLayer(big1, big0, 2048, 256, 512, 68, 69, 70);
  runLayer(big0, h_l3, 2048, 512, 1024, 71, 72, 73);

  // --- fp1: known=unknown=l2 (nk=nu=128), kfeat=l3(1024), ufeat=l2_feat(512) ---
  k_fp<<<BATCH * 16, 256, 0, stream>>>(f_l2x, f_l2x, h_l3, 1024, 1024,
                                       h_f2, 512, 512, 128, 128, 1536, big0, 16);
  runLayer(big0, big1, 2048, 1536, 256, 2, 3, 4);
  runLayer(big1, h_up1, 2048, 256, 256, 5, 6, 7);

  // --- fp2: known=l2(128), unknown=l1(512), kfeat=up1(256), ufeat=l1_feat(320) ---
  k_fp<<<BATCH * 64, 256, 0, stream>>>(f_l2x, f_l1x, h_up1, 256, 256,
                                       h_f1, 320, 320, 128, 512, 576, big0, 64);
  runLayer(big0, big1, 8192, 576, 256, 8, 9, 10);
  runLayer(big1, h_up2, 8192, 256, 128, 11, 12, 13);

  // --- fp3: known=l1(512), unknown=xyz(8192), kfeat=up2(128), ufeat=l0(22) ---
  k_l0<<<gsBlocks((size_t)BATCH * NPTS * 22), 256, 0, stream>>>(oneh, f_xyz, h_norm, h_l0);
  k_fp<<<BATCH * 1024, 256, 0, stream>>>(f_l1x, f_xyz, h_up2, 128, 128,
                                         h_l0, 22, 22, 512, NPTS, 160, big0, 1024);
  runLayer(big0, big1, 131072, 160, 128, 14, 15, 16);
  runLayer(big1, h_up3, 131072, 128, 128, 17, 18, 19);

  // --- final transpose to (B, 128, N) f32 ---
  k_out<<<gsBlocks((size_t)BATCH * 128 * NPTS), 256, 0, stream>>>(h_up3, (float*)d_out);
}
